// GCNClassifier_25907242730199
// MI455X (gfx1250) — compile-verified, matrix operand order checked by reading, not run on hardware
//
#include <hip/hip_runtime.h>
#include <hip/hip_bf16.h>

// ---------------------------------------------------------------------------
// GCN/SAGE classifier for MI455X (gfx1250), bf16 WMMA GEMM core.
// ---------------------------------------------------------------------------

#define N_NODES 20000
#define N_EDGES 160000
#define N_GRAPH 32
#define F_INF   67
#define HID     2048
#define PW      1024
#define NCLS    18
#define EPSV    1e-5f
#define SLOPE   0.01f

typedef __attribute__((ext_vector_type(16))) __bf16  v16bf;
typedef __attribute__((ext_vector_type(8)))  float   v8f;

__device__ __forceinline__ unsigned short f32_to_bf16(float f) {
    unsigned int u = __float_as_uint(f);
    u += 0x7FFFu + ((u >> 16) & 1u);          // round-to-nearest-even
    return (unsigned short)(u >> 16);
}

__device__ __forceinline__ unsigned int pack2_bf16(float lo, float hi) {
    return (unsigned int)f32_to_bf16(lo) | ((unsigned int)f32_to_bf16(hi) << 16);
}

// ---------------------------------------------------------------------------
// Generic fill
// ---------------------------------------------------------------------------
__global__ void k_fill(float* __restrict__ p, int n, float v) {
    int i = blockIdx.x * blockDim.x + threadIdx.x;
    if (i < n) p[i] = v;
}

// in-degree of dst nodes
__global__ void k_degree(const int* __restrict__ dst, float* __restrict__ deg, int e) {
    int i = blockIdx.x * blockDim.x + threadIdx.x;
    if (i < e) atomicAdd(&deg[dst[i]], 1.0f);
}

// nsum[dst[e], f] += x[src[e], f]
__global__ void k_edge_scatter(const float* __restrict__ x,
                               const int* __restrict__ src,
                               const int* __restrict__ dst,
                               float* __restrict__ nsum, int e, int f) {
    int i = blockIdx.x * blockDim.x + threadIdx.x;
    if (i >= e * f) return;
    int ei = i / f, fi = i - ei * f;
    atomicAdd(&nsum[(size_t)dst[ei] * f + fi], x[(size_t)src[ei] * f + fi]);
}

// nsum[i,f] /= max(deg[i],1)
__global__ void k_scale_rows(float* __restrict__ nsum, const float* __restrict__ deg,
                             int n, int f) {
    int i = blockIdx.x * blockDim.x + threadIdx.x;
    if (i >= n * f) return;
    int r = i / f;
    float d = deg[r];
    nsum[i] *= 1.0f / fmaxf(d, 1.0f);
}

// ---------------------------------------------------------------------------
// Weight packing: W = [W1 ; W2] (K1+K2 x Nc, row-major) -> bf16 fragment order.
// Fragment (nt, kt): 32 lanes x 16 bf16. Lane l<16 holds col n=nt*16+l with
// K offsets {0..7,16..23}; lane l>=16 holds same col set with {8..15,24..31}.
// ---------------------------------------------------------------------------
__global__ void k_packW(const float* __restrict__ W1, int K1,
                        const float* __restrict__ W2, int K2,
                        int Nc, int KT, int NT,
                        unsigned short* __restrict__ Wp) {
    int e = blockIdx.x * blockDim.x + threadIdx.x;
    int total = NT * KT * 512;
    if (e >= total) return;
    int nt  = e / (KT * 512);
    int rem = e - nt * (KT * 512);
    int kt  = rem >> 9;
    int r2  = rem & 511;
    int ln  = r2 >> 4;
    int j   = r2 & 15;
    int n   = nt * 16 + (ln & 15);
    int grp = j >> 3, jj = j & 7;
    int kin = (ln < 16) ? (grp * 16 + jj) : (8 + grp * 16 + jj);
    int kk  = kt * 32 + kin;
    float v = 0.0f;
    if (n < Nc) {
        if (kk < K1)           v = W1[(size_t)kk * Nc + n];
        else if (kk < K1 + K2) v = W2[(size_t)(kk - K1) * Nc + n];
    }
    Wp[e] = f32_to_bf16(v);
}

// ---------------------------------------------------------------------------
// FAST bf16 WMMA GEMM, block tile 128(M) x 256(N):
//   requires K1%32==0, K2%32==0, Nc%256==0; grid=(Nc/256, ceil(M/128))
// 256 threads = 8 wave32; wave w owns 16-row strip w, computes 16 WMMAs/K-step
// against 16 B fragments staged in LDS (shared by all 8 waves). A rows are
// clamped (min(row,M-1)) so the K-loop is branch-free; stores are guarded.
// Per-K-step LDS: A = 8 frags (8KB), B = 16 frags (16KB).
// ---------------------------------------------------------------------------
__global__ __launch_bounds__(256)
void k_gemm_wmma_fast(const float* __restrict__ A1, int K1,
                      const float* __restrict__ A2, int K2,
                      const unsigned short* __restrict__ Wp,
                      const float* __restrict__ bias,
                      float* __restrict__ Cout,
                      int M, int Nc, int KT, int epi) {
    __shared__ __attribute__((aligned(32))) unsigned short sA[8 * 512];
    __shared__ __attribute__((aligned(32))) unsigned short sB[16 * 512];

    const int tid  = threadIdx.x;
    const int lane = tid & 31;
    const int wave = tid >> 5;
    const int mBlk   = blockIdx.y * 128;
    const int ntBase = blockIdx.x * 16;

    // --- A fill slot: thread t fills lane (t&31) of fragment (t>>5) (16 bf16)
    const int fa_frag = tid >> 5;                 // == wave
    const int fa_ln   = tid & 31;
    const int fa_g0   = (fa_ln < 16) ? 0 : 8;     // K-group base within tile
    int fa_row = mBlk + fa_frag * 16 + (fa_ln & 15);
    fa_row = min(fa_row, M - 1);                  // clamp: branch-free loads
    unsigned int* sAw = (unsigned int*)(sA + (size_t)tid * 16);

    // --- B fill slot: 16 threads per fragment, 32 ushorts (64B) each
    const int fb_frag = tid >> 4;                 // 0..15 local n-tile
    const int fb_off  = (tid & 15) * 32;          // ushort offset within frag
    uint4* sBw = (uint4*)(sB + (size_t)fb_frag * 512 + fb_off);
    const size_t bStride = (size_t)KT * 512;      // ushorts per n-tile

    v8f acc[16];
    #pragma unroll
    for (int j = 0; j < 16; ++j) acc[j] = (v8f)0.0f;

    for (int kt = 0; kt < KT; ++kt) {
        // ---- cooperative A fragment fill (fp32 -> bf16, vectorized) ----
        {
            int kk = kt * 32 + fa_g0;
            const float* ap = (kk < K1) ? (A1 + (size_t)fa_row * K1 + kk)
                                        : (A2 + (size_t)fa_row * K2 + (kk - K1));
            float4 f0 = *(const float4*)(ap);
            float4 f1 = *(const float4*)(ap + 4);
            float4 f2 = *(const float4*)(ap + 16);
            float4 f3 = *(const float4*)(ap + 20);
            sAw[0] = pack2_bf16(f0.x, f0.y);
            sAw[1] = pack2_bf16(f0.z, f0.w);
            sAw[2] = pack2_bf16(f1.x, f1.y);
            sAw[3] = pack2_bf16(f1.z, f1.w);
            sAw[4] = pack2_bf16(f2.x, f2.y);
            sAw[5] = pack2_bf16(f2.z, f2.w);
            sAw[6] = pack2_bf16(f3.x, f3.y);
            sAw[7] = pack2_bf16(f3.z, f3.w);
        }
        // ---- cooperative B fragment staging (prepacked bf16, 128b copies) ----
        {
            const unsigned short* gp =
                Wp + (size_t)(ntBase + fb_frag) * bStride + (size_t)kt * 512 + fb_off;
            const uint4* gv = (const uint4*)gp;
            if (kt + 1 < KT) __builtin_prefetch(gp + 512, 0, 1);
            sBw[0] = gv[0];
            sBw[1] = gv[1];
            sBw[2] = gv[2];
            sBw[3] = gv[3];
        }
        __syncthreads();

        v16bf a = *(const v16bf*)(sA + (size_t)wave * 512 + (size_t)lane * 16);
        #pragma unroll
        for (int j = 0; j < 16; ++j) {
            v16bf b = *(const v16bf*)(sB + (size_t)j * 512 + (size_t)lane * 16);
            acc[j] = __builtin_amdgcn_wmma_f32_16x16x32_bf16(
                false, a, false, b, (short)0, acc[j], false, false);
        }
        __syncthreads();
    }

    const int rowBase = mBlk + wave * 16 + ((lane >= 16) ? 8 : 0);
    const int colLane = lane & 15;
    #pragma unroll
    for (int j = 0; j < 16; ++j) {
        int col = (ntBase + j) * 16 + colLane;
        float bv = bias[col];
        #pragma unroll
        for (int r = 0; r < 8; ++r) {
            int row = rowBase + r;
            if (row < M) {
                float v = acc[j][r] + bv;
                if (epi == 1) v = (v >= 0.0f) ? v : SLOPE * v;
                Cout[(size_t)row * Nc + col] = v;
            }
        }
    }
}

// ---------------------------------------------------------------------------
// Generic (guarded) bf16 WMMA GEMM for odd shapes (layer-1 K=134, fc3 Nc=18).
// Block tile 32x256, 8 waves = 2 M-subtiles x 4 N-subtiles.
// ---------------------------------------------------------------------------
__global__ __launch_bounds__(256)
void k_gemm_wmma(const float* __restrict__ A1, int K1,
                 const float* __restrict__ A2, int K2,
                 const unsigned short* __restrict__ Wp,
                 const float* __restrict__ bias,
                 float* __restrict__ Cout,
                 int M, int Nc, int NT, int KT, int epi) {
    __shared__ __attribute__((aligned(32))) unsigned short sA[2][32][16];

    const int lane   = threadIdx.x & 31;
    const int wave   = threadIdx.x >> 5;
    const int msub   = wave >> 2;
    const int nsub   = wave & 3;
    const int mBlk   = blockIdx.y * 32;
    const int ntBase = blockIdx.x * 16 + nsub * 4;

    v8f acc[4];
    #pragma unroll
    for (int j = 0; j < 4; ++j) acc[j] = (v8f)0.0f;

    for (int kt = 0; kt < KT; ++kt) {
        #pragma unroll
        for (int i = 0; i < 4; ++i) {
            int e    = threadIdx.x * 4 + i;
            int ms   = e >> 9;
            int ln   = (e >> 4) & 31;
            int j    = e & 15;
            int grp  = j >> 3, jj = j & 7;
            int kin  = (ln < 16) ? (grp * 16 + jj) : (8 + grp * 16 + jj);
            int row  = mBlk + ms * 16 + (ln & 15);
            int kk   = kt * 32 + kin;
            float v  = 0.0f;
            if (row < M) {
                if (kk < K1)           v = A1[(size_t)row * K1 + kk];
                else if (kk < K1 + K2) v = A2[(size_t)row * K2 + (kk - K1)];
            }
            ((unsigned short*)sA)[e] = f32_to_bf16(v);
        }
        __syncthreads();

        v16bf a = *(const v16bf*)(&sA[msub][lane][0]);

        #pragma unroll
        for (int j = 0; j < 4; ++j) {
            int nt = ntBase + j;                 // wave-uniform guard
            if (nt < NT) {
                const unsigned short* bp =
                    Wp + (((size_t)nt * KT + kt) * 32 + lane) * 16;
                v16bf b = *(const v16bf*)bp;
                acc[j] = __builtin_amdgcn_wmma_f32_16x16x32_bf16(
                    false, a, false, b, (short)0, acc[j], false, false);
            }
        }
        __syncthreads();
    }

    const int rowBase = mBlk + msub * 16 + ((lane >= 16) ? 8 : 0);
    const int colLane = lane & 15;
    #pragma unroll
    for (int j = 0; j < 4; ++j) {
        int nt = ntBase + j;
        if (nt >= NT) continue;
        int col = nt * 16 + colLane;
        if (col >= Nc) continue;
        float bv = bias ? bias[col] : 0.0f;
        #pragma unroll
        for (int r = 0; r < 8; ++r) {
            int row = rowBase + r;
            if (row < M) {
                float v = acc[j][r] + bv;
                if (epi == 1) v = (v >= 0.0f) ? v : SLOPE * v;
                Cout[(size_t)row * Nc + col] = v;
            }
        }
    }
}

// ---------------------------------------------------------------------------
// BatchNorm: column partial sums -> finalize -> apply(+leaky)
// ---------------------------------------------------------------------------
__global__ void k_bn_partial(const float* __restrict__ C, int M, int Nc,
                             float* __restrict__ sums, float* __restrict__ sumsq) {
    int col = blockIdx.x * blockDim.x + threadIdx.x;
    if (col >= Nc) return;
    int per = (M + gridDim.y - 1) / gridDim.y;
    int r0 = blockIdx.y * per;
    int r1 = min(M, r0 + per);
    float s = 0.0f, s2 = 0.0f;
    for (int r = r0; r < r1; ++r) {
        float v = C[(size_t)r * Nc + col];
        s += v; s2 += v * v;
    }
    atomicAdd(&sums[col], s);
    atomicAdd(&sumsq[col], s2);
}

__global__ void k_bn_final(const float* __restrict__ sums, const float* __restrict__ sumsq,
                           float* __restrict__ mean, float* __restrict__ var,
                           int M, int Nc) {
    int col = blockIdx.x * blockDim.x + threadIdx.x;
    if (col >= Nc) return;
    float m = sums[col] / (float)M;
    mean[col] = m;
    var[col]  = sumsq[col] / (float)M - m * m;
}

__global__ void k_bn_apply(const float* __restrict__ C,
                           const float* __restrict__ mean, const float* __restrict__ var,
                           const float* __restrict__ gamma, const float* __restrict__ beta,
                           float* __restrict__ out, int n, int Nc) {
    int i = blockIdx.x * blockDim.x + threadIdx.x;
    if (i >= n) return;
    int col = i % Nc;
    float v = (C[i] - mean[col]) * rsqrtf(var[col] + EPSV) * gamma[col] + beta[col];
    out[i] = (v >= 0.0f) ? v : SLOPE * v;
}

// ---------------------------------------------------------------------------
// Graph mean pooling
// ---------------------------------------------------------------------------
__global__ void k_pool_scatter(const float* __restrict__ x, const int* __restrict__ gids,
                               float* __restrict__ gsum, int n, int f) {
    int i = blockIdx.x * blockDim.x + threadIdx.x;
    if (i >= n * f) return;
    int r = i / f, fi = i - r * f;
    atomicAdd(&gsum[(size_t)gids[r] * f + fi], x[i]);
}

__global__ void k_pool_cnt(const int* __restrict__ gids, float* __restrict__ cnt, int n) {
    int i = blockIdx.x * blockDim.x + threadIdx.x;
    if (i < n) atomicAdd(&cnt[gids[i]], 1.0f);
}

__global__ void k_pool_final(const float* __restrict__ gsum, const float* __restrict__ cnt,
                             float* __restrict__ hg, int g, int f) {
    int i = blockIdx.x * blockDim.x + threadIdx.x;
    if (i >= g * f) return;
    int r = i / f;
    hg[i] = gsum[i] / fmaxf(cnt[r], 1.0f);
}

// ---------------------------------------------------------------------------
// Host orchestration
// ---------------------------------------------------------------------------
static inline size_t alignUp(size_t x, size_t a) { return (x + a - 1) & ~(a - 1); }

extern "C" void kernel_launch(void* const* d_in, const int* in_sizes, int n_in,
                              void* d_out, int out_size, void* d_ws, size_t ws_size,
                              hipStream_t stream) {
    const float* h    = (const float*)d_in[0];
    const int*   src  = (const int*)d_in[1];
    const int*   dst  = (const int*)d_in[2];
    const int*   gids = (const int*)d_in[3];
    const float* ws1 = (const float*)d_in[4];  const float* wn1 = (const float*)d_in[5];
    const float* b1  = (const float*)d_in[6];
    const float* ws2 = (const float*)d_in[7];  const float* wn2 = (const float*)d_in[8];
    const float* b2  = (const float*)d_in[9];
    const float* ws3 = (const float*)d_in[10]; const float* wn3 = (const float*)d_in[11];
    const float* b3  = (const float*)d_in[12];
    const float* g1  = (const float*)d_in[13]; const float* be1 = (const float*)d_in[14];
    const float* g2  = (const float*)d_in[15]; const float* be2 = (const float*)d_in[16];
    const float* g3  = (const float*)d_in[17]; const float* be3 = (const float*)d_in[18];
    const float* fw1 = (const float*)d_in[19]; const float* fb1 = (const float*)d_in[20];
    const float* fw2 = (const float*)d_in[21]; const float* fb2 = (const float*)d_in[22];
    const float* fw3 = (const float*)d_in[23]; const float* fb3 = (const float*)d_in[24];

    // ---- workspace carve ----
    char* base = (char*)d_ws;
    size_t off = 0;
    auto carve = [&](size_t bytes) -> char* {
        char* p = base + off;
        off += alignUp(bytes, 256);
        return p;
    };
    float* nsum = (float*)carve((size_t)N_NODES * HID * 4);
    float* act  = (float*)carve((size_t)N_NODES * HID * 4);
    float* Cbuf = (float*)carve((size_t)N_NODES * HID * 4);
    unsigned short* Wp = (unsigned short*)carve((size_t)4096 * 2048 * 2);
    float* deg   = (float*)carve((size_t)N_NODES * 4);
    float* sums  = (float*)carve((size_t)HID * 4);
    float* sumsq = (float*)carve((size_t)HID * 4);
    float* meanb = (float*)carve((size_t)HID * 4);
    float* varb  = (float*)carve((size_t)HID * 4);
    float* gsum  = (float*)carve((size_t)N_GRAPH * HID * 4);
    float* cnt   = (float*)carve((size_t)N_GRAPH * 4);
    float* hg    = (float*)carve((size_t)N_GRAPH * HID * 4);
    float* y1    = (float*)carve((size_t)N_GRAPH * HID * 4);
    float* y2    = (float*)carve((size_t)N_GRAPH * PW * 4);

    auto blocks = [](long n) { return (unsigned)((n + 255) / 256); };

    // in-degree (reused by all three layers)
    k_fill<<<blocks(N_NODES), 256, 0, stream>>>(deg, N_NODES, 0.0f);
    k_degree<<<blocks(N_EDGES), 256, 0, stream>>>(dst, deg, N_EDGES);

    // generic (guarded) GEMM for odd shapes
    auto runGemm = [&](const float* A1, int K1, const float* A2, int K2,
                       const float* W1, const float* W2,
                       const float* bias, float* Cout, int M, int Nc, int epi) {
        int KT = (K1 + K2 + 31) / 32;
        int NT = (Nc + 15) / 16;
        long pel = (long)NT * KT * 512;
        k_packW<<<blocks(pel), 256, 0, stream>>>(W1, K1, W2, K2, Nc, KT, NT, Wp);
        dim3 grid((NT + 15) / 16, (M + 31) / 32);
        k_gemm_wmma<<<grid, 256, 0, stream>>>(A1, K1, A2, K2, Wp, bias, Cout,
                                              M, Nc, NT, KT, epi);
    };

    // fast GEMM: K1%32==0, K2%32==0, Nc%256==0 (block tile 128x256)
    auto runGemmFast = [&](const float* A1, int K1, const float* A2, int K2,
                           const float* W1, const float* W2,
                           const float* bias, float* Cout, int M, int Nc, int epi) {
        int KT = (K1 + K2) / 32;
        int NT = Nc / 16;
        long pel = (long)NT * KT * 512;
        k_packW<<<blocks(pel), 256, 0, stream>>>(W1, K1, W2, K2, Nc, KT, NT, Wp);
        dim3 grid(Nc / 256, (unsigned)((M + 127) / 128));
        k_gemm_wmma_fast<<<grid, 256, 0, stream>>>(A1, K1, A2, K2, Wp, bias, Cout,
                                                   M, Nc, KT, epi);
    };

    auto runBN = [&](const float* Cin, const float* gamma, const float* beta,
                     float* out) {
        k_fill<<<blocks(HID), 256, 0, stream>>>(sums, HID, 0.0f);
        k_fill<<<blocks(HID), 256, 0, stream>>>(sumsq, HID, 0.0f);
        dim3 g(blocks(HID), 40);
        k_bn_partial<<<g, 256, 0, stream>>>(Cin, N_NODES, HID, sums, sumsq);
        k_bn_final<<<blocks(HID), 256, 0, stream>>>(sums, sumsq, meanb, varb, N_NODES, HID);
        k_bn_apply<<<blocks((long)N_NODES * HID), 256, 0, stream>>>(
            Cin, meanb, varb, gamma, beta, out, N_NODES * HID, HID);
    };

    auto runAgg = [&](const float* x, int F) {
        k_fill<<<blocks((long)N_NODES * F), 256, 0, stream>>>(nsum, N_NODES * F, 0.0f);
        k_edge_scatter<<<blocks((long)N_EDGES * F), 256, 0, stream>>>(x, src, dst, nsum,
                                                                      N_EDGES, F);
        k_scale_rows<<<blocks((long)N_NODES * F), 256, 0, stream>>>(nsum, deg, N_NODES, F);
    };

    // ---- layer 1 (K = 67+67, unaligned -> generic path) ----
    runAgg(h, F_INF);
    runGemm(h, F_INF, nsum, F_INF, ws1, wn1, b1, Cbuf, N_NODES, HID, 0);
    runBN(Cbuf, g1, be1, act);

    // ---- layer 2 (K = 2048+2048 -> 128 K-tiles, fast path) ----
    runAgg(act, HID);
    runGemmFast(act, HID, nsum, HID, ws2, wn2, b2, Cbuf, N_NODES, HID, 0);
    runBN(Cbuf, g2, be2, act);

    // ---- layer 3 ----
    runAgg(act, HID);
    runGemmFast(act, HID, nsum, HID, ws3, wn3, b3, Cbuf, N_NODES, HID, 0);
    runBN(Cbuf, g3, be3, act);

    // ---- graph mean pooling ----
    k_fill<<<blocks((long)N_GRAPH * HID), 256, 0, stream>>>(gsum, N_GRAPH * HID, 0.0f);
    k_fill<<<1, 256, 0, stream>>>(cnt, N_GRAPH, 0.0f);
    k_pool_scatter<<<blocks((long)N_NODES * HID), 256, 0, stream>>>(act, gids, gsum,
                                                                    N_NODES, HID);
    k_pool_cnt<<<blocks(N_NODES), 256, 0, stream>>>(gids, cnt, N_NODES);
    k_pool_final<<<blocks((long)N_GRAPH * HID), 256, 0, stream>>>(gsum, cnt, hg,
                                                                  N_GRAPH, HID);

    // ---- MLP head (tiny M=32 rows, still WMMA) ----
    runGemmFast(hg, HID, nullptr, 0, fw1, nullptr, fb1, y1, N_GRAPH, HID, 1);
    runGemmFast(y1, HID, nullptr, 0, fw2, nullptr, fb2, y2, N_GRAPH, PW, 1);
    runGemm(y2, PW, nullptr, 0, fw3, nullptr, fb3, (float*)d_out, N_GRAPH, NCLS, 0);
}
